// ContextualAttention_Enhance_25512105738318
// MI455X (gfx1250) — matrix operand order repaired and checked by reading, not compiled, not run on hardware
//
#include <hip/hip_runtime.h>
#include <hip/hip_bf16.h>

#define T_ 4
#define C_ 64
#define CI_ 16
#define H_ 96
#define W_ 96
#define NPIX (H_*W_)            // 9216
#define PS_ 7
#define WS_ 21
#define NCAND (WS_*WS_)         // 441
#define NQX 24
#define NQ (NQX*NQX)            // 576
#define KS_ 100
#define SCALE_ 10.0f

// score-kernel geometry
#define TILES1D 6
#define NTILES (TILES1D*TILES1D)  // 36 query tiles per frame (4x4 queries each)
#define GRID_ 33                  // union candidate grid 33x33
#define NGRID (GRID_*GRID_)       // 1089
#define TPW 9                     // M tiles per wave (8 waves * 9 = 72 tiles = 1152 rows)
#define KSTEPS 25                 // 49 patch cells -> 25 pairs (last padded)
#define B3R 39
#define B1R 19

typedef __attribute__((ext_vector_type(16))) __bf16 v16bf;
typedef __attribute__((ext_vector_type(8)))  float  v8f;

union FragAB { v16bf v; uint4 q[2]; };

// ---------------------------------------------------------------- zero
__global__ void k_zero(float* __restrict__ p, int n) {
    int i = blockIdx.x * blockDim.x + threadIdx.x;
    if (i < n) p[i] = 0.0f;
}

// ---------------------------------------------------------------- conv1x1 x3
// b1 (g)   -> bf16, channel-last [t][y][x][ci]  (WMMA B operand source)
// b3 (phi) -> bf16, channel-last [t][y][x][ci]  (WMMA A operand source)
// b2 (theta)-> f32, [t][ci][y][x]               (aggregation values)
__global__ void k_conv_in(const float* __restrict__ vid,
                          const float* __restrict__ g_w,  const float* __restrict__ g_b,
                          const float* __restrict__ th_w, const float* __restrict__ th_b,
                          const float* __restrict__ ph_w, const float* __restrict__ ph_b,
                          __bf16* __restrict__ b1h, __bf16* __restrict__ b3h,
                          float* __restrict__ b2f) {
    int idx = blockIdx.x * blockDim.x + threadIdx.x;
    if (idx >= T_ * NPIX) return;
    int t = idx / NPIX, p = idx - t * NPIX;
    float v[C_];
    const float* src = vid + (size_t)t * C_ * NPIX + p;
    #pragma unroll 8
    for (int c = 0; c < C_; ++c) v[c] = src[(size_t)c * NPIX];

    __bf16* d1 = b1h + (size_t)idx * CI_;
    __bf16* d3 = b3h + (size_t)idx * CI_;
    for (int o = 0; o < CI_; ++o) {
        float a1 = g_b[o], a2 = th_b[o], a3 = ph_b[o];
        const float* w1 = g_w  + o * C_;
        const float* w2 = th_w + o * C_;
        const float* w3 = ph_w + o * C_;
        #pragma unroll 8
        for (int c = 0; c < C_; ++c) {
            a1 += w1[c] * v[c];
            a2 += w2[c] * v[c];
            a3 += w3[c] * v[c];
        }
        d1[o] = (__bf16)a1;
        d3[o] = (__bf16)a3;
        b2f[((size_t)t * CI_ + o) * NPIX + p] = a2;
    }
}

// ---------------------------------------------------------------- WMMA scores
// One workgroup per (t, 4x4 query tile). GEMM D[1089 cand, 16 query], K'=800
// (k' = patchcell*16 + channel, zero-padded), bf16 inputs, f32 accumulate.
// LDS regions are staged with border-clamped data, so inner-loop addressing is
// clamp-free: per-tile base + compile-time per-k-step immediate offset.
__global__ __launch_bounds__(256) void k_scores(const __bf16* __restrict__ b1h,
                                                const __bf16* __restrict__ b3h,
                                                float* __restrict__ dists) {
    __shared__ alignas(16) __bf16 b3lds[B3R * B3R * CI_];  // 48672 B
    __shared__ alignas(16) __bf16 b1lds[B1R * B1R * CI_];  // 11552 B

    int blk = blockIdx.x;
    int t  = blk / NTILES, tl = blk - t * NTILES;
    int ty = tl / TILES1D, tx = tl - ty * TILES1D;
    int qy0 = 16 * ty, qx0 = 16 * tx;
    int tid = threadIdx.x;

    // stage key region (rows/cols clamped at the borders): rows qy0-10 .. qy0+28
    for (int cell = tid; cell < B3R * B3R; cell += 256) {
        int r = cell / B3R, c = cell - r * B3R;
        int gy = min(max(qy0 - 10 + r, 0), H_ - 1);
        int gx = min(max(qx0 - 10 + c, 0), W_ - 1);
        const uint4* s4 = (const uint4*)(b3h + ((size_t)t * NPIX + gy * W_ + gx) * CI_);
        uint4* d4 = (uint4*)(b3lds + cell * CI_);
        d4[0] = s4[0]; d4[1] = s4[1];
    }
    // stage query region: rows qy0 .. qy0+18 (bottom/right clamped)
    for (int cell = tid; cell < B1R * B1R; cell += 256) {
        int r = cell / B1R, c = cell - r * B1R;
        int gy = min(qy0 + r, H_ - 1);
        int gx = min(qx0 + c, W_ - 1);
        const uint4* s4 = (const uint4*)(b1h + ((size_t)t * NPIX + gy * W_ + gx) * CI_);
        uint4* d4 = (uint4*)(b1lds + cell * CI_);
        d4[0] = s4[0]; d4[1] = s4[1];
    }
    __syncthreads();

    int lane = tid & 31, wid = tid >> 5;
    int m16  = lane & 15;        // A: M within tile / D: N index
    int half = lane >> 4;        // K half select for A; M+8 select for D
    int iyq  = m16 >> 2, ixq = m16 & 3;   // query position within tile (for B/D)

    // per-(lane,tile) base offset (elements) into b3lds; clamp-free inner loop
    int tb[TPW];
    #pragma unroll
    for (int i = 0; i < TPW; ++i) {
        int m = (wid * TPW + i) * 16 + m16;
        if (m > NGRID - 1) m = NGRID - 1;   // pad rows: harmless garbage
        int ay = m / GRID_, ax = m - ay * GRID_;
        int cy = min(max(qy0 - 10 + ay, 0), H_ - 1);   // clamped candidate pos
        int cx = min(max(qx0 - 10 + ax, 0), W_ - 1);
        tb[i] = (((cy - qy0 + 10) * B3R) + (cx - qx0 + 10)) * CI_ + half * 8;
    }
    // per-lane base offset into b1lds for the B operand
    int qb = ((4 * iyq) * B1R + 4 * ixq) * CI_;

    v8f acc[TPW];
    #pragma unroll
    for (int i = 0; i < TPW; ++i)
        #pragma unroll
        for (int e = 0; e < 8; ++e) acc[i][e] = 0.0f;

    #pragma unroll
    for (int s = 0; s < KSTEPS; ++s) {
        const int c0 = 2 * s, c1 = 2 * s + 1;
        const int py0 = c0 / 7, px0 = c0 - py0 * 7;
        const int py1 = c1 / 7, px1 = c1 - py1 * 7;
        const bool c1ok = (c1 < 49);
        const int offA0 = (py0 * B3R + px0) * CI_;     // compile-time constants
        const int offA1 = (py1 * B3R + px1) * CI_;
        const int offB0 = (py0 * B1R + px0) * CI_;
        const int offB1 = (py1 * B1R + px1) * CI_;

        // B fragment: V0..3 = cell c0 (ci half), V4..7 = cell c1 (ci half)
        FragAB B;
        B.q[0] = *(const uint4*)(b1lds + (qb + offB0 + half * 8));
        if (c1ok) B.q[1] = *(const uint4*)(b1lds + (qb + offB1 + half * 8));
        else      B.q[1] = make_uint4(0, 0, 0, 0);

        #pragma unroll
        for (int i = 0; i < TPW; ++i) {
            // A fragment: V0..3 = cell c0 (ci half), V4..7 = cell c1 (ci half)
            FragAB A;
            A.q[0] = *(const uint4*)(b3lds + (tb[i] + offA0));
            if (c1ok) A.q[1] = *(const uint4*)(b3lds + (tb[i] + offA1));
            else      A.q[1] = make_uint4(0, 0, 0, 0);
            acc[i] = __builtin_amdgcn_wmma_f32_16x16x32_bf16(
                false, A.v, false, B.v, (short)0, acc[i], false, false);
        }
    }

    // D layout: lanes 0-15: N=lane, VGPR v -> M=v ; lanes 16-31: M=v+8
    int qg = (4 * ty + iyq) * NQX + (4 * tx + ixq);
    #pragma unroll
    for (int i = 0; i < TPW; ++i) {
        int tile = wid * TPW + i;
        #pragma unroll
        for (int v = 0; v < 8; ++v) {
            int m = tile * 16 + v + 8 * half;
            if (m >= NGRID) continue;
            int ay = m / GRID_, ax = m - ay * GRID_;
            int oy = ay - 4 * iyq, ox = ax - 4 * ixq;
            if ((unsigned)oy < WS_ && (unsigned)ox < WS_) {
                dists[((size_t)t * NQ + qg) * NCAND + oy * WS_ + ox] = acc[i][v];
            }
        }
    }
}

// ---------------------------------------------------------------- top-k + softmax + aggregate
// One wave (32 lanes) per (t, query). Greedy 100x wave-argmax == top_k set,
// online softmax (first pick is the max), fused weighted patch accumulation.
__global__ __launch_bounds__(32) void k_attend(const float* __restrict__ dists,
                                               const float* __restrict__ b2f,
                                               float* __restrict__ accb,
                                               float* __restrict__ Zb) {
    int blk = blockIdx.x;                  // t*NQ + qg
    int t = blk / NQ, qg = blk - t * NQ;
    int lane = threadIdx.x;
    int qy = (qg / NQX) * 4, qx = (qg % NQX) * 4;

    const float* dq = dists + (size_t)blk * NCAND;
    float v[14];
    #pragma unroll
    for (int i = 0; i < 14; ++i) {
        int j = lane + 32 * i;
        v[i] = (j < NCAND) ? dq[j] : -1e30f;
    }

    // per-lane decomposition of d = lane + 32*i  (d = ci*49 + py*7 + px)
    int ciA[25], pyA[25], pxA[25];
    bool okA[25];
    #pragma unroll
    for (int i = 0; i < 25; ++i) {
        int d = lane + 32 * i;
        okA[i] = (d < CI_ * 49);
        int dd = okA[i] ? d : 0;
        int ci = dd / 49, r = dd - ci * 49;
        int py = r / 7;
        ciA[i] = ci; pyA[i] = py; pxA[i] = r - py * 7;
    }
    float z[25];
    #pragma unroll
    for (int i = 0; i < 25; ++i) z[i] = 0.0f;

    float vmax = 0.0f, esum = 0.0f;
    for (int k = 0; k < KS_; ++k) {
        float lm = -1e30f; int li = 0x7FFFFFFF;
        #pragma unroll
        for (int i = 0; i < 14; ++i)
            if (v[i] > lm) { lm = v[i]; li = lane + 32 * i; }
        #pragma unroll
        for (int off = 16; off > 0; off >>= 1) {
            float ov = __shfl_xor(lm, off, 32);
            int   oi = __shfl_xor(li, off, 32);
            if (ov > lm || (ov == lm && oi < li)) { lm = ov; li = oi; }
        }
        if (k == 0) vmax = lm;
        float e = __expf((lm - vmax) * SCALE_);
        esum += e;
        // invalidate winner entry (owner lane only)
        int ii = li - lane;
        if (ii >= 0 && (ii & 31) == 0) {
            int s = ii >> 5;
            if (s < 14) v[s] = -1e30f;
        }
        int oy = li / WS_, ox = li - oy * WS_;
        int cy = min(max(qy + oy - 10, 0), H_ - 1);
        int cx = min(max(qx + ox - 10, 0), W_ - 1);
        #pragma unroll
        for (int i = 0; i < 25; ++i) {
            if (!okA[i]) continue;
            int gy = min(cy + pyA[i], H_ - 1);
            int gx = min(cx + pxA[i], W_ - 1);
            z[i] += e * b2f[((size_t)t * CI_ + ciA[i]) * NPIX + gy * W_ + gx];
        }
    }

    float rinv = 1.0f / esum;
    #pragma unroll
    for (int i = 0; i < 25; ++i) {
        if (!okA[i]) continue;
        int gy = min(qy + pyA[i], H_ - 1);
        int gx = min(qx + pxA[i], W_ - 1);
        atomicAdd(&accb[((size_t)t * CI_ + ciA[i]) * NPIX + gy * W_ + gx], z[i] * rinv);
        if (t == 0 && (lane + 32 * i) < 49)           // one count per (q,py,px)
            atomicAdd(&Zb[gy * W_ + gx], 1.0f);
    }
}

// ---------------------------------------------------------------- out = vid + conv1x1(acc/Z)
__global__ void k_out(const float* __restrict__ vid,
                      const float* __restrict__ accb, const float* __restrict__ Zb,
                      const float* __restrict__ W_w,  const float* __restrict__ W_b,
                      float* __restrict__ out) {
    int idx = blockIdx.x * blockDim.x + threadIdx.x;
    if (idx >= T_ * NPIX) return;
    int t = idx / NPIX, p = idx - t * NPIX;
    float zinv = 1.0f / Zb[p];
    float y16[CI_];
    #pragma unroll
    for (int ci = 0; ci < CI_; ++ci)
        y16[ci] = accb[((size_t)t * CI_ + ci) * NPIX + p] * zinv;
    for (int c = 0; c < C_; ++c) {
        float a = W_b[c];
        const float* w = W_w + c * CI_;
        #pragma unroll
        for (int ci = 0; ci < CI_; ++ci) a += w[ci] * y16[ci];
        out[((size_t)t * C_ + c) * NPIX + p] =
            vid[((size_t)t * C_ + c) * NPIX + p] + a;
    }
}

// ---------------------------------------------------------------- launch
extern "C" void kernel_launch(void* const* d_in, const int* in_sizes, int n_in,
                              void* d_out, int out_size, void* d_ws, size_t ws_size,
                              hipStream_t stream) {
    const float* vid  = (const float*)d_in[0];
    const float* g_w  = (const float*)d_in[1];
    const float* g_b  = (const float*)d_in[2];
    const float* th_w = (const float*)d_in[3];
    const float* th_b = (const float*)d_in[4];
    const float* ph_w = (const float*)d_in[5];
    const float* ph_b = (const float*)d_in[6];
    const float* W_w  = (const float*)d_in[7];
    const float* W_b  = (const float*)d_in[8];
    float* out = (float*)d_out;

    char* p = (char*)d_ws;
    auto carve = [&](size_t bytes) -> char* {
        char* r = p;
        p += (bytes + 255) & ~(size_t)255;
        return r;
    };
    __bf16* b1h  = (__bf16*)carve((size_t)2 * T_ * NPIX * CI_);
    __bf16* b3h  = (__bf16*)carve((size_t)2 * T_ * NPIX * CI_);
    float*  b2f  = (float*) carve((size_t)4 * T_ * NPIX * CI_);
    float*  dsts = (float*) carve((size_t)4 * T_ * NQ * NCAND);
    float*  accb = (float*) carve((size_t)4 * T_ * NPIX * CI_);
    float*  Zb   = (float*) carve((size_t)4 * NPIX);

    int nz = T_ * NPIX * CI_;
    k_zero<<<(nz + 255) / 256, 256, 0, stream>>>(accb, nz);
    k_zero<<<(NPIX + 255) / 256, 256, 0, stream>>>(Zb, NPIX);
    k_conv_in<<<(T_ * NPIX + 255) / 256, 256, 0, stream>>>(
        vid, g_w, g_b, th_w, th_b, ph_w, ph_b, b1h, b3h, b2f);
    k_scores<<<T_ * NTILES, 256, 0, stream>>>(b1h, b3h, dsts);
    k_attend<<<T_ * NQ, 32, 0, stream>>>(dsts, b2f, accb, Zb);
    k_out<<<(T_ * NPIX + 255) / 256, 256, 0, stream>>>(vid, accb, Zb, W_w, W_b, out);
}
